// RNNSS_Labeler_32452772888626
// MI455X (gfx1250) — matrix operand order repaired and checked by reading, not compiled, not run on hardware
//
#include <hip/hip_runtime.h>
#include <hip/hip_bf16.h>

// ---------------------------------------------------------------------------
// RNN sequence labeler for MI455X (gfx1250).
//  * Persistent kernel, 128 WGs x 256 threads, runs all 4096 timesteps with
//    4 grid barriers/step (embed -> gates/LSTM -> interm -> tag/softmax).
//  * Weights pre-converted to bf16 in workspace (56.7 MB -> L2-resident on
//    the 192 MB L2); each weight byte read exactly once per step.
//  * GEMV via v_wmma_f32_16x16x32_bf16: A = activation vector broadcast into
//    all 16 rows (bf16 copy in LDS, k = 16h + s per lane half), B = 16 weight rows
//    (one contiguous 32-byte global load per lane giving k = 16h + s, as A).
//    acc[0] of each lane = dot(x, W[row0 + lane&15]) with f32 accumulation.
//  * Gates computed one at a time to keep a single live accumulator
//    (reduces VGPR pressure -> no spills, better residency for grid sync).
// ---------------------------------------------------------------------------

typedef __attribute__((ext_vector_type(16))) __bf16    v16bf;
typedef __attribute__((ext_vector_type(8)))  float     v8f;
typedef __attribute__((ext_vector_type(8)))  unsigned  v8u;

#define NWG      128
#define TSTEPS   4096
#define EMBED    1024
#define HIDDEN   2048
#define INTERM   1024
#define TAGSET   16
#define INPUTD   1024
#define CATD     1040   // INPUTD + TAGSET (fp32 W_embed row pitch)

__device__ __forceinline__ unsigned short f32_to_bf16(float f) {
  unsigned u = __float_as_uint(f);
  unsigned r = u + 0x7FFFu + ((u >> 16) & 1u);   // round-to-nearest-even
  return (unsigned short)(r >> 16);
}

__device__ __forceinline__ v8f wmma_bf16(v16bf a, v16bf b, v8f c) {
  return __builtin_amdgcn_wmma_f32_16x16x32_bf16(
      /*neg_a=*/false, a, /*neg_b=*/false, b,
      /*c_mod=*/(short)0, c, /*reuse_a=*/false, /*reuse_b=*/false);
}

// Accumulate dot-products for 16 output rows [row0, row0+16) over K-chunks
// [c0, c1) of 32. packLDS holds the activation vector as bf16 in natural
// k order (k = 16h + s per lane half h): per chunk c, halves [c*32 .. c*32+15] are the
// low-lane block (k = 0..15) and [c*32+16 .. c*32+31] the high-lane
// block (k = 16..31).
__device__ __forceinline__ v8f gemv_accum(v8f acc,
                                          const unsigned short* __restrict__ Wbf,
                                          int row0, int pitch,
                                          const unsigned short* packLDS,
                                          int c0, int c1, int lane) {
  const int hi = lane >> 4;                         // 0: K 0..15, 1: K 16..31
  const unsigned short* wrow =
      Wbf + (size_t)(row0 + (lane & 15)) * pitch + hi * 16;
  const unsigned short* ap = packLDS + hi * 16;
  for (int c = c0; c < c1; ++c) {
    __builtin_prefetch(wrow + (size_t)(c + 1) * 32, 0, 1);  // global_prefetch_b8
    v8u braw = *reinterpret_cast<const v8u*>(wrow + (size_t)c * 32);
    v8u araw = *reinterpret_cast<const v8u*>(ap + c * 32);
    acc = wmma_bf16(__builtin_bit_cast(v16bf, araw),
                    __builtin_bit_cast(v16bf, braw), acc);
  }
  return acc;
}

// Copy an activation vector into LDS as bf16 in natural order (pack[j] = vec[j]).
__device__ __forceinline__ void build_pack(const float* __restrict__ vec,
                                           unsigned short* pack, int n) {
  for (int j = threadIdx.x; j < n; j += blockDim.x) {
    int c = j >> 5, jj = j & 31, g = jj >> 3, o = jj & 7;
    int k = (c << 5) + jj;
    pack[j] = f32_to_bf16(vec[k]);
  }
}

__device__ __forceinline__ void grid_sync(unsigned* ctrl) {
  __syncthreads();
  if (threadIdx.x == 0) {
    unsigned* cnt = ctrl + 0;
    unsigned* gen = ctrl + 1;
    unsigned g = __hip_atomic_load(gen, __ATOMIC_RELAXED, __HIP_MEMORY_SCOPE_AGENT);
    unsigned v = __hip_atomic_fetch_add(cnt, 1u, __ATOMIC_ACQ_REL, __HIP_MEMORY_SCOPE_AGENT);
    if (v == NWG - 1u) {
      __hip_atomic_store(cnt, 0u, __ATOMIC_RELAXED, __HIP_MEMORY_SCOPE_AGENT);
      __hip_atomic_store(gen, g + 1u, __ATOMIC_RELEASE, __HIP_MEMORY_SCOPE_AGENT);
    } else {
      while (__hip_atomic_load(gen, __ATOMIC_ACQUIRE, __HIP_MEMORY_SCOPE_AGENT) == g)
        __builtin_amdgcn_s_sleep(1);
    }
  }
  __syncthreads();
}

__device__ __forceinline__ float sigm(float x) { return 1.f / (1.f + expf(-x)); }

// --------------------------- utility kernels -------------------------------

__global__ void k_convert_bf16(const float* __restrict__ src,
                               unsigned short* __restrict__ dst,
                               long long rows, long long cols, long long spitch) {
  long long total = rows * cols;
  long long stride = (long long)gridDim.x * blockDim.x;
  for (long long i = (long long)blockIdx.x * blockDim.x + threadIdx.x;
       i < total; i += stride) {
    long long r = i / cols, c = i - r * cols;
    dst[i] = f32_to_bf16(src[r * spitch + c]);
  }
}

__global__ void k_init(unsigned* ctrl, float* h0, float* cst) {
  int i = blockIdx.x * blockDim.x + threadIdx.x;
  if (i < 8) ctrl[i] = 0u;                 // cnt, gen, pred
  if (i < HIDDEN) { h0[i] = 0.f; cst[i] = 0.f; }
}

// --------------------------- persistent kernel -----------------------------

__global__ void __launch_bounds__(256, 2)
rnn_persistent(const float* __restrict__ seq,
               const float* __restrict__ Wemb_f, const float* __restrict__ b_emb,
               const float* __restrict__ b_ih,   const float* __restrict__ b_hh,
               const float* __restrict__ b_int,  const float* __restrict__ b_tag,
               const unsigned short* __restrict__ We,
               const unsigned short* __restrict__ Wih,
               const unsigned short* __restrict__ Whh,
               const unsigned short* __restrict__ Wint,
               const unsigned short* __restrict__ Wtag,
               float* e_g, float* h0_g, float* h1_g, float* c_g, float* inter_g,
               unsigned* ctrl,
               float* out_logp, float* out_prob, int* out_pred) {
  __shared__ alignas(32) unsigned short pack1[1024];   // x_t / e / inter
  __shared__ alignas(32) unsigned short pack2[2048];   // h
  __shared__ float red[8][64];
  __shared__ float ts[16];

  const int wg = blockIdx.x;
  const int tid = threadIdx.x;
  const int wv = tid >> 5;
  const int lane = tid & 31;

  for (int t = 0; t < TSTEPS; ++t) {
    const float* hprev = (t & 1) ? h1_g : h0_g;
    float*       hnew  = (t & 1) ? h0_g : h1_g;

    // ---- Phase A: e = relu(W_embed[:, :1024] @ x_t + col(pred) + b) -------
    if (wg < EMBED / 16) {
      build_pack(seq + (size_t)t * INPUTD, pack1, INPUTD);
      __syncthreads();
      v8f acc = {};
      acc = gemv_accum(acc, We, wg * 16, INPUTD, pack1, wv * 4, wv * 4 + 4, lane);
      if (lane < 16) red[wv][lane] = acc[0];
      __syncthreads();
      if (tid < 16) {
        int r = wg * 16 + tid;
        float s = b_emb[r];
        for (int k = 0; k < 8; ++k) s += red[k][tid];
        unsigned p = __hip_atomic_load(ctrl + 2, __ATOMIC_ACQUIRE,
                                       __HIP_MEMORY_SCOPE_AGENT);
        s += Wemb_f[(size_t)r * CATD + INPUTD + (int)p];   // one-hot column
        e_g[r] = s > 0.f ? s : 0.f;
      }
    }
    grid_sync(ctrl);

    // ---- Phase B: gates = Wih@e + Whh@h + b; LSTM cell update -------------
    build_pack(e_g, pack1, EMBED);
    build_pack(hprev, pack2, HIDDEN);
    __syncthreads();
    // One gate at a time: single live accumulator keeps VGPR pressure low.
    #pragma unroll 1
    for (int g = 0; g < 4; ++g) {
      int row = g * HIDDEN + wg * 16;
      v8f acc = {};
      acc = gemv_accum(acc, Wih, row, EMBED,  pack1, wv * 4, wv * 4 + 4, lane);
      acc = gemv_accum(acc, Whh, row, HIDDEN, pack2, wv * 8, wv * 8 + 8, lane);
      if (lane < 16) red[wv][g * 16 + lane] = acc[0];
    }
    __syncthreads();
    if (tid < 16) {
      int r = wg * 16 + tid;
      float s[4];
      for (int g = 0; g < 4; ++g) {
        float v = b_ih[g * HIDDEN + r] + b_hh[g * HIDDEN + r];
        for (int k = 0; k < 8; ++k) v += red[k][g * 16 + tid];
        s[g] = v;
      }
      float cn = sigm(s[1]) * c_g[r] + sigm(s[0]) * tanhf(s[2]);
      c_g[r] = cn;
      hnew[r] = sigm(s[3]) * tanhf(cn);
    }
    grid_sync(ctrl);

    // ---- Phase C: inter = relu(W_interm @ h + b) --------------------------
    if (wg < INTERM / 16) {
      build_pack(hnew, pack2, HIDDEN);
      __syncthreads();
      v8f acc = {};
      acc = gemv_accum(acc, Wint, wg * 16, HIDDEN, pack2, wv * 8, wv * 8 + 8, lane);
      if (lane < 16) red[wv][lane] = acc[0];
      __syncthreads();
      if (tid < 16) {
        int r = wg * 16 + tid;
        float s = b_int[r];
        for (int k = 0; k < 8; ++k) s += red[k][tid];
        inter_g[r] = s > 0.f ? s : 0.f;
      }
    }
    grid_sync(ctrl);

    // ---- Phase D: tag_space, log_softmax, softmax, argmax (WG 0) ----------
    if (wg == 0) {
      build_pack(inter_g, pack1, INTERM);
      __syncthreads();
      v8f acc = {};
      acc = gemv_accum(acc, Wtag, 0, INTERM, pack1, wv * 4, wv * 4 + 4, lane);
      if (lane < 16) red[wv][lane] = acc[0];
      __syncthreads();
      if (tid < 16) {
        float s = b_tag[tid];
        for (int k = 0; k < 8; ++k) s += red[k][tid];
        ts[tid] = s;
      }
      __syncthreads();
      if (tid == 0) {
        float m = ts[0]; int am = 0;
        for (int j = 1; j < 16; ++j) if (ts[j] > m) { m = ts[j]; am = j; }
        float se = 0.f;
        for (int j = 0; j < 16; ++j) se += expf(ts[j] - m);
        float lz = m + logf(se);
        for (int j = 0; j < 16; ++j) {
          float lp = ts[j] - lz;
          out_logp[t * 16 + j] = lp;
          out_prob[t * 16 + j] = expf(lp);
        }
        out_pred[t] = am;
        __hip_atomic_store(ctrl + 2, (unsigned)am, __ATOMIC_RELEASE,
                           __HIP_MEMORY_SCOPE_AGENT);
      }
    }
    grid_sync(ctrl);
  }
}

// ------------------------------- launcher ----------------------------------

extern "C" void kernel_launch(void* const* d_in, const int* in_sizes, int n_in,
                              void* d_out, int out_size, void* d_ws, size_t ws_size,
                              hipStream_t stream) {
  (void)in_sizes; (void)n_in; (void)out_size; (void)ws_size;

  const float* seq    = (const float*)d_in[0];
  const float* Wemb   = (const float*)d_in[1];
  const float* b_emb  = (const float*)d_in[2];
  const float* Wih    = (const float*)d_in[3];
  const float* Whh    = (const float*)d_in[4];
  const float* b_ih   = (const float*)d_in[5];
  const float* b_hh   = (const float*)d_in[6];
  const float* Wint   = (const float*)d_in[7];
  const float* b_int  = (const float*)d_in[8];
  const float* Wtag   = (const float*)d_in[9];
  const float* b_tag  = (const float*)d_in[10];

  // carve workspace
  char* p = (char*)d_ws;
  auto carve = [&](size_t bytes) -> void* {
    void* r = (void*)p;
    p += (bytes + 255) & ~(size_t)255;
    return r;
  };
  unsigned* ctrl      = (unsigned*)carve(64);
  float* e_g          = (float*)carve(EMBED * 4);
  float* h0_g         = (float*)carve(HIDDEN * 4);
  float* h1_g         = (float*)carve(HIDDEN * 4);
  float* c_g          = (float*)carve(HIDDEN * 4);
  float* inter_g      = (float*)carve(INTERM * 4);
  unsigned short* We_bf   = (unsigned short*)carve((size_t)EMBED * INPUTD * 2);
  unsigned short* Wih_bf  = (unsigned short*)carve((size_t)4 * HIDDEN * EMBED * 2);
  unsigned short* Whh_bf  = (unsigned short*)carve((size_t)4 * HIDDEN * HIDDEN * 2);
  unsigned short* Wint_bf = (unsigned short*)carve((size_t)INTERM * HIDDEN * 2);
  unsigned short* Wtag_bf = (unsigned short*)carve((size_t)TAGSET * INTERM * 2);

  k_init<<<8, 256, 0, stream>>>(ctrl, h0_g, c_g);
  k_convert_bf16<<<2048, 256, 0, stream>>>(Wemb, We_bf,  EMBED,      INPUTD, CATD);
  k_convert_bf16<<<4096, 256, 0, stream>>>(Wih,  Wih_bf, 4 * HIDDEN, EMBED,  EMBED);
  k_convert_bf16<<<4096, 256, 0, stream>>>(Whh,  Whh_bf, 4 * HIDDEN, HIDDEN, HIDDEN);
  k_convert_bf16<<<2048, 256, 0, stream>>>(Wint, Wint_bf, INTERM,    HIDDEN, HIDDEN);
  k_convert_bf16<<<64,   256, 0, stream>>>(Wtag, Wtag_bf, TAGSET,    INTERM, INTERM);

  float* out_logp = (float*)d_out;
  float* out_prob = out_logp + (size_t)TSTEPS * TAGSET;
  int*   out_pred = (int*)(out_logp + (size_t)2 * TSTEPS * TAGSET);

  rnn_persistent<<<NWG, 256, 0, stream>>>(
      seq, Wemb, b_emb, b_ih, b_hh, b_int, b_tag,
      We_bf, Wih_bf, Whh_bf, Wint_bf, Wtag_bf,
      e_g, h0_g, h1_g, c_g, inter_g, ctrl,
      out_logp, out_prob, out_pred);
}
